// LocalSlidingWindowAttention_31086973288777
// MI455X (gfx1250) — compile-verified, matrix operand order checked by reading, not run on hardware
//
#include <hip/hip_runtime.h>
#include <hip/hip_bf16.h>

typedef _Float16 half_t;
typedef __attribute__((ext_vector_type(16))) _Float16 v16h;
typedef __attribute__((ext_vector_type(8)))  float    v8f;
typedef __attribute__((ext_vector_type(4)))  int      v4i;

#define D_MODEL 768
#define L_SEQ   4096
#define HEADS   12
#define HDIM    64
#define HALFWIN 64   // WINDOW/2

#if defined(__has_builtin)
#if __has_builtin(__builtin_amdgcn_global_load_async_to_lds_b128)
#define HAVE_ASYNC_LDS 1
#endif
#endif

__device__ __forceinline__ int clampi(int x, int lo, int hi) {
    return x < lo ? lo : (x > hi ? hi : x);
}

#ifdef HAVE_ASYNC_LDS
typedef __attribute__((address_space(1))) v4i* gptr_v4i;
typedef __attribute__((address_space(3))) v4i* lptr_v4i;
__device__ __forceinline__ void async_cp16(const half_t* g, half_t* l) {
    __builtin_amdgcn_global_load_async_to_lds_b128(
        (gptr_v4i)g, (lptr_v4i)l, 0, 0);
}
#endif

// ---------------------------------------------------------------------------
// one-shot f32 -> f16 conversion (inputs + weights), L2-resident
// ---------------------------------------------------------------------------
__global__ __launch_bounds__(256) void cvt_f32_f16(
    const float* __restrict__ s, half_t* __restrict__ d, int n)
{
    const int i = (blockIdx.x * 256 + threadIdx.x) * 8;
    if (i < n) {
#pragma unroll
        for (int j = 0; j < 8; ++j) d[i + j] = (half_t)s[i + j];
    }
}

// ---------------------------------------------------------------------------
// GEMM: C[M,N] = X[M,K] * W^T + b   (X, W both f16; W is [N,K] row-major)
// Block tile 128x64, 8 waves (4x2), wave tile 32x32 -> 4 WMMA / K-step.
// ---------------------------------------------------------------------------
template <typename TOUT>
__global__ __launch_bounds__(256) void gemm_xwt_f16(
    const half_t* __restrict__ X, const half_t* __restrict__ W,
    const float* __restrict__ bias, TOUT* __restrict__ C,
    int M, int N, int K)
{
    __shared__ half_t As[128][40];   // X tile  (rows m, cols k), padded
    __shared__ half_t Bs[64][40];    // W tile  (rows n, cols k), padded

    const int tid  = threadIdx.x;
    const int wave = tid >> 5;
    const int lane = tid & 31;
    const int m0 = blockIdx.x * 128;
    const int n0 = blockIdx.y * 64;
    const int wm = (wave & 3) * 32;        // wave row offset in block tile
    const int wn = (wave >> 2) * 32;       // wave col offset in block tile
    const int kb = (lane >> 4) * 8;        // K striping base for fragments

    v8f acc[2][2] = {};

    for (int k0 = 0; k0 < K; k0 += 32) {
        // ---- stage tiles into LDS ----
#ifdef HAVE_ASYNC_LDS
        {
#pragma unroll
            for (int t = 0; t < 2; ++t) {
                const int c   = tid + t * 256;     // 0..511
                const int row = c >> 2;
                const int col = (c & 3) * 8;
                async_cp16(X + (size_t)(m0 + row) * K + k0 + col, &As[row][col]);
            }
            const int brow = tid >> 2;             // 0..63
            const int bcol = (tid & 3) * 8;
            async_cp16(W + (size_t)(n0 + brow) * K + k0 + bcol, &Bs[brow][bcol]);
            asm volatile("s_wait_asynccnt 0" ::: "memory");
        }
#else
        {
            const int ar = tid >> 1;               // 0..127
            const int ac = (tid & 1) * 16;
            const half_t* src = X + (size_t)(m0 + ar) * K + k0 + ac;
#pragma unroll
            for (int j = 0; j < 16; ++j) As[ar][ac + j] = src[j];
            const int br = tid >> 2;               // 0..63
            const int bc = (tid & 3) * 8;
            const half_t* wsrc = W + (size_t)(n0 + br) * K + k0 + bc;
#pragma unroll
            for (int j = 0; j < 8; ++j) Bs[br][bc + j] = wsrc[j];
        }
#endif
        __syncthreads();

        // ---- fragments ----
        v16h a[2], b[2];
#pragma unroll
        for (int mi = 0; mi < 2; ++mi) {
            const int m = wm + mi * 16 + (lane & 15);
#pragma unroll
            for (int j = 0; j < 8; ++j) {
                a[mi][j]     = As[m][kb + j];
                a[mi][8 + j] = As[m][kb + 16 + j];
            }
        }
#pragma unroll
        for (int ni = 0; ni < 2; ++ni) {
            const int n = wn + ni * 16 + (lane & 15);
#pragma unroll
            for (int j = 0; j < 8; ++j) {
                b[ni][j]     = Bs[n][kb + j];
                b[ni][8 + j] = Bs[n][kb + 16 + j];
            }
        }
#pragma unroll
        for (int mi = 0; mi < 2; ++mi)
#pragma unroll
            for (int ni = 0; ni < 2; ++ni)
                acc[mi][ni] = __builtin_amdgcn_wmma_f32_16x16x32_f16(
                    false, a[mi], false, b[ni], (short)0, acc[mi][ni], false, false);
        __syncthreads();
    }

    // ---- writeback: lane%16 = column, rows = v + 8*(lane/16) ----
#pragma unroll
    for (int mi = 0; mi < 2; ++mi)
#pragma unroll
        for (int ni = 0; ni < 2; ++ni) {
            const int col = n0 + wn + ni * 16 + (lane & 15);
            const float bvv = bias[col];
            const int rbase = m0 + wm + mi * 16 + 8 * (lane >> 4);
#pragma unroll
            for (int v = 0; v < 8; ++v)
                C[(size_t)(rbase + v) * N + col] = (TOUT)(acc[mi][ni][v] + bvv);
        }
}

// ---------------------------------------------------------------------------
// Sliding-window attention: one wave per (head, 16-query block).
// Key span for a 16-query block: [q0-64, q0+79] -> 10 tiles of 16 (160 keys).
// ---------------------------------------------------------------------------
__global__ __launch_bounds__(128) void attn(
    const half_t* __restrict__ Q, const half_t* __restrict__ Kb,
    const half_t* __restrict__ Vb, half_t* __restrict__ O)
{
    __shared__ half_t Ps[4][16][176];      // per-wave P staging (f16)

    const int wave = threadIdx.x >> 5;
    const int lane = threadIdx.x & 31;
    const int task = blockIdx.x * 4 + wave;        // 0..3071
    const int h    = task >> 8;                    // head
    const int q0   = (task & 255) * 16;            // query block start
    const int colb = h * HDIM;                     // column base in [L,768]
    const int kb   = (lane >> 4) * 8;              // K striping base

    // ---- load Q fragments (16x64 A-matrix = two 16x32 fragments) ----
    v16h aQ0, aQ1;
    {
        const half_t* qrow = Q + (size_t)(q0 + (lane & 15)) * D_MODEL + colb;
#pragma unroll
        for (int j = 0; j < 8; ++j) {
            aQ0[j]     = qrow[kb + j];
            aQ0[8 + j] = qrow[kb + 16 + j];
            aQ1[j]     = qrow[32 + kb + j];
            aQ1[8 + j] = qrow[32 + kb + 16 + j];
        }
    }

    // ---- scores for 10 key tiles ----
    v8f   s[10];
    float rmax[8];
#pragma unroll
    for (int v = 0; v < 8; ++v) rmax[v] = -1e30f;

#pragma unroll
    for (int t = 0; t < 10; ++t) {
        const int kstart = q0 - HALFWIN + t * 16;
        const int key    = kstart + (lane & 15);
        const int keyc   = clampi(key, 0, L_SEQ - 1);
        const half_t* krow = Kb + (size_t)keyc * D_MODEL + colb;
        v16h b0, b1;
#pragma unroll
        for (int j = 0; j < 8; ++j) {
            b0[j]     = krow[kb + j];
            b0[8 + j] = krow[kb + 16 + j];
            b1[j]     = krow[32 + kb + j];
            b1[8 + j] = krow[32 + kb + 16 + j];
        }
        v8f c = {};
        c = __builtin_amdgcn_wmma_f32_16x16x32_f16(false, aQ0, false, b0,
                                                   (short)0, c, false, false);
        c = __builtin_amdgcn_wmma_f32_16x16x32_f16(false, aQ1, false, b1,
                                                   (short)0, c, false, false);
#pragma unroll
        for (int v = 0; v < 8; ++v) {
            const int row  = q0 + v + 8 * (lane >> 4);
            const int dif  = row - key;
            const bool ok  = (key >= 0) & (key < L_SEQ) &
                             (dif <= HALFWIN) & (dif >= -HALFWIN);
            const float sv = ok ? c[v] * 0.125f : -1e30f;   // 1/sqrt(64)
            s[t][v] = sv;
            rmax[v] = fmaxf(rmax[v], sv);
        }
    }

    // ---- row max across the 16 lanes of each half-wave ----
#pragma unroll
    for (int v = 0; v < 8; ++v) {
        float x = rmax[v];
#pragma unroll
        for (int off = 1; off < 16; off <<= 1)
            x = fmaxf(x, __shfl_xor(x, off, 32));
        rmax[v] = x;
    }

    // ---- exponentiate, row sum, stage P to LDS ----
    float rsum[8];
#pragma unroll
    for (int v = 0; v < 8; ++v) rsum[v] = 0.f;
#pragma unroll
    for (int t = 0; t < 10; ++t) {
#pragma unroll
        for (int v = 0; v < 8; ++v) {
            const float p = __expf(s[t][v] - rmax[v]);
            rsum[v] += p;
            Ps[wave][v + 8 * (lane >> 4)][t * 16 + (lane & 15)] = (half_t)p;
        }
    }
#pragma unroll
    for (int v = 0; v < 8; ++v) {
        float x = rsum[v];
#pragma unroll
        for (int off = 1; off < 16; off <<= 1)
            x += __shfl_xor(x, off, 32);
        rsum[v] = 1.0f / x;
    }

    // ---- O = P(16x160) * V(160x64), 5 K-steps x 4 N-tiles ----
    v8f o[4] = {{}, {}, {}, {}};
#pragma unroll
    for (int ks = 0; ks < 5; ++ks) {
        const int k2 = ks * 32;
        v16h aP;
        {
            const int m = lane & 15;
#pragma unroll
            for (int j = 0; j < 8; ++j) {
                aP[j]     = Ps[wave][m][k2 + kb + j];
                aP[8 + j] = Ps[wave][m][k2 + kb + 16 + j];
            }
        }
#pragma unroll
        for (int nt = 0; nt < 4; ++nt) {
            const int dcol = colb + nt * 16 + (lane & 15);
            v16h bV;
#pragma unroll
            for (int j = 0; j < 8; ++j) {
                const int key0 = clampi(q0 - HALFWIN + k2 + kb + j,      0, L_SEQ - 1);
                const int key1 = clampi(q0 - HALFWIN + k2 + kb + 16 + j, 0, L_SEQ - 1);
                bV[j]     = Vb[(size_t)key0 * D_MODEL + dcol];
                bV[8 + j] = Vb[(size_t)key1 * D_MODEL + dcol];
            }
            o[nt] = __builtin_amdgcn_wmma_f32_16x16x32_f16(false, aP, false, bV,
                                                           (short)0, o[nt], false, false);
        }
    }

    // ---- normalize + store (f16, row-major [L,768]) ----
#pragma unroll
    for (int nt = 0; nt < 4; ++nt) {
#pragma unroll
        for (int v = 0; v < 8; ++v) {
            const int row = q0 + v + 8 * (lane >> 4);
            const int col = colb + nt * 16 + (lane & 15);
            O[(size_t)row * D_MODEL + col] = (half_t)(o[nt][v] * rsum[v]);
        }
    }
}

// ---------------------------------------------------------------------------
extern "C" void kernel_launch(void* const* d_in, const int* in_sizes, int n_in,
                              void* d_out, int out_size, void* d_ws, size_t ws_size,
                              hipStream_t stream) {
    const float* query = (const float*)d_in[0];
    const float* key   = (const float*)d_in[1];
    const float* value = (const float*)d_in[2];
    const float* Wq = (const float*)d_in[3];
    const float* bq = (const float*)d_in[4];
    const float* Wk = (const float*)d_in[5];
    const float* bk = (const float*)d_in[6];
    const float* Wv = (const float*)d_in[7];
    const float* bv = (const float*)d_in[8];
    const float* Wo = (const float*)d_in[9];
    const float* bo = (const float*)d_in[10];

    const size_t E  = (size_t)L_SEQ * D_MODEL;      // 3145728
    const size_t WE = (size_t)D_MODEL * D_MODEL;    // 589824

    half_t* Xq  = (half_t*)d_ws;
    half_t* Xk  = Xq + E;
    half_t* Xv  = Xk + E;
    half_t* Wqh = Xv + E;
    half_t* Wkh = Wqh + WE;
    half_t* Wvh = Wkh + WE;
    half_t* Woh = Wvh + WE;
    half_t* Qb  = Woh + WE;
    half_t* Kb  = Qb + E;
    half_t* Vb  = Kb + E;
    half_t* Ob  = Vb + E;

    // f32 -> f16 conversion passes (L2-resident)
    cvt_f32_f16<<<(int)(E / 2048), 256, 0, stream>>>(query, Xq, (int)E);
    cvt_f32_f16<<<(int)(E / 2048), 256, 0, stream>>>(key,   Xk, (int)E);
    cvt_f32_f16<<<(int)(E / 2048), 256, 0, stream>>>(value, Xv, (int)E);
    cvt_f32_f16<<<(int)(WE / 2048), 256, 0, stream>>>(Wq, Wqh, (int)WE);
    cvt_f32_f16<<<(int)(WE / 2048), 256, 0, stream>>>(Wk, Wkh, (int)WE);
    cvt_f32_f16<<<(int)(WE / 2048), 256, 0, stream>>>(Wv, Wvh, (int)WE);
    cvt_f32_f16<<<(int)(WE / 2048), 256, 0, stream>>>(Wo, Woh, (int)WE);

    dim3 ggrid(L_SEQ / 128, D_MODEL / 64, 1);
    gemm_xwt_f16<half_t><<<ggrid, 256, 0, stream>>>(Xq, Wqh, bq, Qb,
                                                    L_SEQ, D_MODEL, D_MODEL);
    gemm_xwt_f16<half_t><<<ggrid, 256, 0, stream>>>(Xk, Wkh, bk, Kb,
                                                    L_SEQ, D_MODEL, D_MODEL);
    gemm_xwt_f16<half_t><<<ggrid, 256, 0, stream>>>(Xv, Wvh, bv, Vb,
                                                    L_SEQ, D_MODEL, D_MODEL);

    attn<<<(HEADS * (L_SEQ / 16)) / 4, 128, 0, stream>>>(Qb, Kb, Vb, Ob);

    gemm_xwt_f16<float><<<ggrid, 256, 0, stream>>>(Ob, Woh, bo, (float*)d_out,
                                                   L_SEQ, D_MODEL, D_MODEL);
}
